// Allocator_74534862455188
// MI455X (gfx1250) — matrix operand order repaired and checked by reading, not compiled, not run on hardware
//
#include <hip/hip_runtime.h>
#include <hip/hip_bf16.h>
#include <math.h>

#define D     1024
#define E     8
#define BB    8
#define PP    1024
#define NTOK  (BB*PP)      /* 8192 tokens  */
#define NROWS (NTOK*2)     /* 16384 gathered rows (top-2) */
#define MT    32           /* rows per MLP block tile     */
#define LDSS  1032         /* padded LDS row stride (bf16 elems): 2064B -> 4-bank step, conflict-free */

typedef __attribute__((ext_vector_type(4)))  unsigned int u32x4;
typedef __attribute__((ext_vector_type(16))) __bf16       v16bf;
typedef __attribute__((ext_vector_type(8)))  float        v8f;

union Frag16 { u32x4 u[2]; v16bf v; };

__device__ inline unsigned short f2bf(float f) {
  unsigned int u = __float_as_uint(f);
  u += 0x7fffu + ((u >> 16) & 1u);           // round-to-nearest-even
  return (unsigned short)(u >> 16);
}

__device__ inline v8f wmma_bf16(v16bf a, v16bf b, v8f c) {
  // v_wmma_f32_16x16x32_bf16  D = A(16x32) * B(32x16) + C
  return __builtin_amdgcn_wmma_f32_16x16x32_bf16(false, a, false, b, (short)0, c, false, false);
}

// A fragment (16x32 bf16) from LDS, row-major with stride LDSS.
// Lane l: group g=l>>4, r=l&15 -> row mbase+r; K runs [g*8..g*8+7] and [16+g*8..16+g*8+7].
__device__ inline v16bf load_a_lds(const unsigned short* S, int mbase, int kbase, int lane) {
  int g = lane >> 4, r = lane & 15;
  const unsigned short* p = S + (mbase + r) * LDSS + kbase + g * 8;
  Frag16 f;
  f.u[0] = *(const u32x4*)(p);        // 8 bf16 = 16B
  f.u[1] = *(const u32x4*)(p + 16);   // K+16 run
  return f.v;
}

// B fragment (32x16 bf16) from K-major transposed weights WT[n][k].
// Lane l: c=l&15 -> column nbase+c; g=l>>4 -> K runs [kbase+g*8 .. +7] and [kbase+16+g*8 .. +7].
__device__ inline v16bf load_b_gl(const unsigned short* WT, int nbase, int kbase, int lane) {
  int g = lane >> 4, c = lane & 15;
  const unsigned short* p = WT + (size_t)(nbase + c) * D + kbase + g * 8;
  Frag16 f;
  f.u[0] = *(const u32x4*)(p);
  f.u[1] = *(const u32x4*)(p + 16);
  return f.v;
}

// ---------------------------------------------------------------- gating ----
__global__ __launch_bounds__(256) void k_gate(const float* __restrict__ x,
                                              const float* __restrict__ Wg,
                                              const float* __restrict__ bg,
                                              int* __restrict__ counts,
                                              int* __restrict__ rowlist) {
  int wave = threadIdx.x >> 5, lane = threadIdx.x & 31;
  int tok = blockIdx.x * 8 + wave;
  float acc[E];
#pragma unroll
  for (int e = 0; e < E; ++e) acc[e] = 0.f;
  const float* xr = x + ((size_t)tok << 10);
  for (int d = lane; d < D; d += 32) {
    float xv = xr[d];
    const float* wr = Wg + d * E;
#pragma unroll
    for (int e = 0; e < E; ++e) acc[e] += xv * wr[e];
  }
#pragma unroll
  for (int e = 0; e < E; ++e) {
#pragma unroll
    for (int off = 16; off > 0; off >>= 1) acc[e] += __shfl_xor(acc[e], off, 32);
  }
  if (lane == 0) {
#pragma unroll
    for (int e = 0; e < E; ++e) acc[e] += bg[e];
    // softmax is monotonic -> top-2 on logits; jax tie-break = lowest index
    int i0 = 0; float m0 = acc[0];
#pragma unroll
    for (int e = 1; e < E; ++e) if (acc[e] > m0) { m0 = acc[e]; i0 = e; }
    int i1 = 0; float m1 = -3.4e38f;
#pragma unroll
    for (int e = 0; e < E; ++e) if (e != i0 && acc[e] > m1) { m1 = acc[e]; i1 = e; }
    int p0 = atomicAdd(&counts[i0], 1); rowlist[i0 * NROWS + p0] = 2 * tok;
    int p1 = atomicAdd(&counts[i1], 1); rowlist[i1 * NROWS + p1] = 2 * tok + 1;
  }
}

// ----------------------------------------------- gathered activations bf16 --
__global__ __launch_bounds__(256) void k_xg(const float* __restrict__ x,
                                            const float* __restrict__ noise,
                                            unsigned short* __restrict__ xg) {
  size_t gid  = (size_t)blockIdx.x * 256 + threadIdx.x;
  size_t base = gid * 4;                       // element index into [NROWS, D]
  int row = (int)(base >> 10);
  int col = (int)(base & (D - 1));
  float4 xv = *(const float4*)(x + ((size_t)(row >> 1) << 10) + col);
  float4 nv = *(const float4*)(noise + ((size_t)row << 10) + col);
  unsigned int lo = (unsigned)f2bf(xv.x + nv.x) | ((unsigned)f2bf(xv.y + nv.y) << 16);
  unsigned int hi = (unsigned)f2bf(xv.z + nv.z) | ((unsigned)f2bf(xv.w + nv.w) << 16);
  unsigned int* q = (unsigned int*)(xg + base);
  q[0] = lo; q[1] = hi;
}

// -------------------------------------- weight transpose + cast to bf16 -----
__global__ __launch_bounds__(256) void k_wtrans(const float* __restrict__ W1,
                                                const float* __restrict__ W2,
                                                unsigned short* __restrict__ w1t,
                                                unsigned short* __restrict__ w2t) {
  __shared__ float tile[32][33];
  int blk = blockIdx.x;
  int which = blk >> 13;                 // 8192 blocks per matrix
  int rem = blk & 8191;
  int e = rem >> 10, t = rem & 1023;
  int bi = t >> 5, bj = t & 31;
  const float* sp = (which ? W2 : W1) + (size_t)e * D * D;
  unsigned short* dp = (which ? w2t : w1t) + (size_t)e * D * D;
  int tx = threadIdx.x & 31, ty = threadIdx.x >> 5;
#pragma unroll
  for (int p = 0; p < 4; ++p) {
    int r = bi * 32 + ty + p * 8;
    tile[ty + p * 8][tx] = sp[(size_t)r * D + bj * 32 + tx];
  }
  __syncthreads();
#pragma unroll
  for (int p = 0; p < 4; ++p) {
    int n = bj * 32 + ty + p * 8;        // output "row" = N, inner = K
    dp[(size_t)n * D + bi * 32 + tx] = f2bf(tile[tx][ty + p * 8]);
  }
}

// ----------------------------------------------------- fused expert MLP -----
__global__ __launch_bounds__(256) void k_moe(const unsigned short* __restrict__ xg,
                                             const unsigned short* __restrict__ w1t,
                                             const unsigned short* __restrict__ w2t,
                                             const float* __restrict__ x,
                                             const float* __restrict__ noise,
                                             const float* __restrict__ b1,
                                             const float* __restrict__ b2,
                                             const float* __restrict__ lng,
                                             const float* __restrict__ lnb,
                                             const int* __restrict__ counts,
                                             const int* __restrict__ rowlist,
                                             float* __restrict__ ybuf,
                                             float* __restrict__ outacc) {
  __shared__ __align__(16) unsigned short Xs[MT * LDSS];   // 66KB, later aliased as bacc
  __shared__ __align__(16) unsigned short Hs[MT * LDSS];   // 66KB
  __shared__ float psum[MT][8];
  __shared__ float psq[MT][8];
  __shared__ float rmean[MT];
  __shared__ float rrstd[MT];
  __shared__ int   rowIdxS[MT];
  __shared__ int   rowBatchS[MT];

  int tid = threadIdx.x;
  int e    = blockIdx.x >> 9;            // NROWS/MT = 512 tiles per expert
  int tile = blockIdx.x & 511;
  int cnt  = counts[e];
  int m0   = tile * MT;
  if (m0 >= cnt) return;

  if (tid < MT) {
    int i = m0 + tid;
    int grow = (i < cnt) ? rowlist[e * NROWS + i] : -1;
    rowIdxS[tid]   = grow;
    rowBatchS[tid] = (grow >= 0) ? (grow >> 11) : 0;   // row / (K*P)
  }
  __syncthreads();

  { // stage routed Xg rows into LDS (bf16), zero-pad invalid rows
    int lr = tid >> 3;
    int c0 = (tid & 7) * 128;
    int grow = rowIdxS[lr];
    const u32x4* src = (grow >= 0) ? (const u32x4*)(xg + (size_t)grow * D + c0) : nullptr;
    u32x4* dst = (u32x4*)(Xs + lr * LDSS + c0);
    u32x4 zv = {};
#pragma unroll
    for (int j = 0; j < 16; ++j) dst[j] = src ? src[j] : zv;
  }
  __syncthreads();

  int wave = tid >> 5, lane = tid & 31;
  int cl = lane & 15, gh = lane >> 4;
  const unsigned short* W1e = w1t + (size_t)e * D * D;
  const unsigned short* W2e = w2t + (size_t)e * D * D;
  const float* b1e = b1 + e * D;
  const float* b2e = b2 + e * D;
  const float* lge = lng + e * D;
  const float* lbe = lnb + e * D;

  // ---- stage 1: H = gelu(Xg*W1 + b1), bf16 into LDS ----
#pragma unroll 1
  for (int o = 0; o < 2; ++o) {
    int nb = o * 512 + wave * 64;
    v8f c[2][4] = {};
    for (int k0 = 0; k0 < D; k0 += 32) {
      v16bf a0 = load_a_lds(Xs, 0,  k0, lane);
      v16bf a1 = load_a_lds(Xs, 16, k0, lane);
#pragma unroll
      for (int nf = 0; nf < 4; ++nf) {
        v16bf b = load_b_gl(W1e, nb + nf * 16, k0, lane);
        c[0][nf] = wmma_bf16(a0, b, c[0][nf]);
        c[1][nf] = wmma_bf16(a1, b, c[1][nf]);
      }
    }
#pragma unroll
    for (int mi = 0; mi < 2; ++mi) {
#pragma unroll
      for (int nf = 0; nf < 4; ++nf) {
        int col  = nb + nf * 16 + cl;
        float bias = b1e[col];
        int rowb = mi * 16 + gh * 8;
#pragma unroll
        for (int j = 0; j < 8; ++j) {
          float v = c[mi][nf][j] + bias;
          v = 0.5f * v * (1.0f + erff(v * 0.70710678118654752f));  // exact gelu
          Hs[(rowb + j) * LDSS + col] = f2bf(v);
        }
      }
    }
  }
  __syncthreads();

  float* bacc = (float*)Xs;              // Xg no longer needed: 8x1024 f32 batch accumulators
#pragma unroll
  for (int j = 0; j < 32; ++j) bacc[j * 256 + tid] = 0.0f;

  // ---- stage 2: y = H*W2 + b2 + residual -> ybuf ----
#pragma unroll 1
  for (int o = 0; o < 2; ++o) {
    int nb = o * 512 + wave * 64;
    v8f c[2][4] = {};
    for (int k0 = 0; k0 < D; k0 += 32) {
      v16bf a0 = load_a_lds(Hs, 0,  k0, lane);
      v16bf a1 = load_a_lds(Hs, 16, k0, lane);
#pragma unroll
      for (int nf = 0; nf < 4; ++nf) {
        v16bf b = load_b_gl(W2e, nb + nf * 16, k0, lane);
        c[0][nf] = wmma_bf16(a0, b, c[0][nf]);
        c[1][nf] = wmma_bf16(a1, b, c[1][nf]);
      }
    }
#pragma unroll
    for (int mi = 0; mi < 2; ++mi) {
#pragma unroll
      for (int nf = 0; nf < 4; ++nf) {
        int col  = nb + nf * 16 + cl;
        float bias = b2e[col];
        int rowb = mi * 16 + gh * 8;
#pragma unroll
        for (int j = 0; j < 8; ++j) {
          int row  = rowb + j;
          int grow = rowIdxS[row];
          if (grow >= 0) {
            float res = x[((size_t)(grow >> 1) << 10) + col] +
                        noise[((size_t)grow << 10) + col];
            ybuf[((size_t)grow << 10) + col] = c[mi][nf][j] + bias + res;
          }
        }
      }
    }
  }
  __threadfence();
  __syncthreads();

  // ---- per-row LayerNorm stats ----
  {
    int lr = tid >> 3, seg = tid & 7;
    int grow = rowIdxS[lr];
    float s = 0.f, ss = 0.f;
    if (grow >= 0) {
      const float* yr = ybuf + ((size_t)grow << 10) + seg * 128;
#pragma unroll 4
      for (int j = 0; j < 128; ++j) { float v = yr[j]; s += v; ss += v * v; }
    }
    psum[lr][seg] = s; psq[lr][seg] = ss;
  }
  __syncthreads();
  if (tid < MT) {
    float s = 0.f, ss = 0.f;
#pragma unroll
    for (int j = 0; j < 8; ++j) { s += psum[tid][j]; ss += psq[tid][j]; }
    float m   = s * (1.0f / D);
    float var = ss * (1.0f / D) - m * m;
    rmean[tid] = m;
    rrstd[tid] = (rowIdxS[tid] >= 0) ? rsqrtf(var + 1e-5f) : 0.f;
  }
  __syncthreads();

  // ---- normalize + per-batch pre-reduction (each column owned by one thread) ----
#pragma unroll 1
  for (int k = 0; k < 4; ++k) {
    int c0 = tid + k * 256;
    float g = lge[c0], bb = lbe[c0];
    for (int lr = 0; lr < MT; ++lr) {
      int grow = rowIdxS[lr];
      if (grow < 0) continue;
      float v = ybuf[((size_t)grow << 10) + c0];
      v = (v - rmean[lr]) * rrstd[lr] * g + bb;
      bacc[rowBatchS[lr] * D + c0] += v;     // exclusive owner -> no atomics
    }
  }
  __syncthreads();
#pragma unroll 1
  for (int k = 0; k < 4; ++k) {
    int c0 = tid + k * 256;
#pragma unroll
    for (int b = 0; b < BB; ++b) {
      float v = bacc[b * D + c0];
      if (v != 0.0f) atomicAdd(&outacc[((size_t)(b * E + e) << 10) + c0], v);
    }
  }
}

// ------------------------------------------------------ final group norm ----
__global__ __launch_bounds__(256) void k_gn(const float* __restrict__ outacc,
                                            const float* __restrict__ gng,
                                            const float* __restrict__ gnb,
                                            float* __restrict__ out) {
  __shared__ float ws[8], wss[8], sm[2];
  int row = blockIdx.x, tid = threadIdx.x;
  const float* in = outacc + ((size_t)row << 10);
  float s = 0.f, ss = 0.f;
  for (int c = tid; c < D; c += 256) { float v = in[c]; s += v; ss += v * v; }
#pragma unroll
  for (int off = 16; off > 0; off >>= 1) { s += __shfl_xor(s, off, 32); ss += __shfl_xor(ss, off, 32); }
  if ((tid & 31) == 0) { ws[tid >> 5] = s; wss[tid >> 5] = ss; }
  __syncthreads();
  if (tid == 0) {
    float S = 0.f, SS = 0.f;
    for (int j = 0; j < 8; ++j) { S += ws[j]; SS += wss[j]; }
    float m = S * (1.0f / D), var = SS * (1.0f / D) - m * m;
    sm[0] = m; sm[1] = rsqrtf(var + 1e-5f);
  }
  __syncthreads();
  float m = sm[0], r = sm[1];
  for (int c = tid; c < D; c += 256)
    out[((size_t)row << 10) + c] = (in[c] - m) * r * gng[c] + gnb[c];
}

// ---------------------------------------------------------------------------
extern "C" void kernel_launch(void* const* d_in, const int* in_sizes, int n_in,
                              void* d_out, int out_size, void* d_ws, size_t ws_size,
                              hipStream_t stream) {
  const float* x     = (const float*)d_in[0];
  const float* noise = (const float*)d_in[1];
  const float* Wg    = (const float*)d_in[2];
  const float* bg    = (const float*)d_in[3];
  const float* W1    = (const float*)d_in[4];
  const float* b1    = (const float*)d_in[5];
  const float* W2    = (const float*)d_in[6];
  const float* b2    = (const float*)d_in[7];
  const float* lng   = (const float*)d_in[8];
  const float* lnb   = (const float*)d_in[9];
  const float* gng   = (const float*)d_in[10];
  const float* gnb   = (const float*)d_in[11];

  // workspace carve-out (~135 MB): xg bf16 | W1T bf16 | W2T bf16 | y f32 | acc f32 | counts | rowlist
  unsigned short* xg  = (unsigned short*)d_ws;
  unsigned short* w1t = xg  + (size_t)NROWS * D;
  unsigned short* w2t = w1t + (size_t)E * D * D;
  float* ybuf   = (float*)(w2t + (size_t)E * D * D);
  float* outacc = ybuf + (size_t)NROWS * D;
  int*   counts  = (int*)(outacc + (size_t)BB * E * D);
  int*   rowlist = counts + 32;

  hipMemsetAsync(counts, 0, 32 * sizeof(int), stream);
  hipMemsetAsync(outacc, 0, (size_t)BB * E * D * sizeof(float), stream);

  k_gate  <<<NTOK / 8, 256, 0, stream>>>(x, Wg, bg, counts, rowlist);
  k_xg    <<<(NROWS * (D / 4)) / 256, 256, 0, stream>>>(x, noise, xg);
  k_wtrans<<<2 * E * (D / 32) * (D / 32), 256, 0, stream>>>(W1, W2, w1t, w2t);
  k_moe   <<<E * (NROWS / MT), 256, 0, stream>>>(xg, w1t, w2t, x, noise, b1, b2,
                                                 lng, lnb, counts, rowlist, ybuf, outacc);
  k_gn    <<<BB * E, 256, 0, stream>>>(outacc, gng, gnb, (float*)d_out);
}
